// ST_GNN_Robust_53171695124554
// MI455X (gfx1250) — compile-verified, matrix operand order checked by reading, not run on hardware
//
#include <hip/hip_runtime.h>
#include <math.h>

typedef _Float16 h16;
typedef _Float16 v16h __attribute__((ext_vector_type(16)));
typedef _Float16 v8h  __attribute__((ext_vector_type(8)));
typedef float    v8f  __attribute__((ext_vector_type(8)));

#define ED 16
#define MONO_NEG_INF 0x007FFFFFu   // encodes -inf in monotonic-uint float order

static __device__ __forceinline__ unsigned f2mono(float f) {
    unsigned u = __float_as_uint(f);
    return (u & 0x80000000u) ? ~u : (u | 0x80000000u);
}
static __device__ __forceinline__ float mono2f(unsigned u) {
    return (u & 0x80000000u) ? __uint_as_float(u & 0x7fffffffu) : __uint_as_float(~u);
}
static __device__ __forceinline__ v16h cat8(v8h lo, v8h hi) {
    return __builtin_shufflevector(lo, hi, 0,1,2,3,4,5,6,7,8,9,10,11,12,13,14,15);
}
static __device__ __forceinline__ v8f wmma16(v16h a, v16h b, v8f c) {
    return __builtin_amdgcn_wmma_f32_16x16x32_f16(false, a, false, b, (short)0, c, false, false);
}

// ---------------- utility kernels ----------------

__global__ void k_fill_u32(unsigned* __restrict__ p, unsigned v, long long n) {
    long long i = (long long)blockIdx.x * blockDim.x + threadIdx.x;
    if (i < n) p[i] = v;
}

// out[n*Kp+k] = trans ? in[k*Ncols+n] : in[n*K+k]   (zero-padded for k>=K)
__global__ void k_tconv(const float* __restrict__ in, h16* __restrict__ out,
                        int K, int Ncols, int Kp, int trans) {
    long long idx = (long long)blockIdx.x * blockDim.x + threadIdx.x;
    long long total = (long long)Ncols * Kp;
    if (idx >= total) return;
    int n = (int)(idx / Kp), k = (int)(idx % Kp);
    float v = 0.0f;
    if (k < K) v = trans ? in[(long long)k * Ncols + n] : in[(long long)n * K + k];
    out[idx] = (h16)v;
}

__global__ void k_eamean(const float* __restrict__ ea, float* __restrict__ out, int E) {
    __shared__ float s[256][ED];
    float acc[ED];
    for (int k = 0; k < ED; ++k) acc[k] = 0.0f;
    for (int e = threadIdx.x; e < E; e += 256) {
        const float* p = ea + (long long)e * ED;
        for (int k = 0; k < ED; ++k) acc[k] += p[k];
    }
    for (int k = 0; k < ED; ++k) s[threadIdx.x][k] = acc[k];
    __syncthreads();
    for (int off = 128; off > 0; off >>= 1) {
        if ((int)threadIdx.x < off)
            for (int k = 0; k < ED; ++k) s[threadIdx.x][k] += s[threadIdx.x + off][k];
        __syncthreads();
    }
    if (threadIdx.x < ED) out[threadIdx.x] = s[0][threadIdx.x] / (float)E;
}

// ---------------- generic WMMA GEMM: C[M,N] = A[M,K] @ Wt^T + bias ----------------
// Wt is f16, stored as [N][K] (i.e. transposed B). K % 32 == 0, N % 64 == 0.
// One wave per block; wave computes a 16(M) x 64(N) strip. A fragment reused 4x.
__global__ void k_gemm(const float* __restrict__ A, const h16* __restrict__ Wt,
                       const float* __restrict__ bias, float* __restrict__ C,
                       int M, int K, int N) {
    const int lane = threadIdx.x;
    const int p = lane & 15, half = lane >> 4;
    const int m0 = blockIdx.x << 4;
    const int n0 = blockIdx.y << 6;
    int mr = m0 + p; if (mr >= M) mr = M - 1;
    const float* arow = A + (long long)mr * K;
    v8f acc[4];
    for (int t = 0; t < 4; ++t)
        for (int r = 0; r < 8; ++r) acc[t][r] = 0.0f;
    for (int k0 = 0; k0 < K; k0 += 32) {
        const float* ap = arow + k0 + half * 8;
        float4 f0 = *(const float4*)(ap);
        float4 f1 = *(const float4*)(ap + 4);
        float4 f2 = *(const float4*)(ap + 16);
        float4 f3 = *(const float4*)(ap + 20);
        v16h a;
        a[0]=(h16)f0.x; a[1]=(h16)f0.y; a[2]=(h16)f0.z; a[3]=(h16)f0.w;
        a[4]=(h16)f1.x; a[5]=(h16)f1.y; a[6]=(h16)f1.z; a[7]=(h16)f1.w;
        a[8]=(h16)f2.x; a[9]=(h16)f2.y; a[10]=(h16)f2.z; a[11]=(h16)f2.w;
        a[12]=(h16)f3.x; a[13]=(h16)f3.y; a[14]=(h16)f3.z; a[15]=(h16)f3.w;
        #pragma unroll
        for (int t = 0; t < 4; ++t) {
            const h16* wp = Wt + (long long)(n0 + t * 16 + p) * K + k0 + half * 8;
            v16h b = cat8(*(const v8h*)wp, *(const v8h*)(wp + 16));
            acc[t] = wmma16(a, b, acc[t]);
        }
    }
    #pragma unroll
    for (int t = 0; t < 4; ++t) {
        int col = n0 + t * 16 + p;
        float bs = bias ? bias[col] : 0.0f;
        #pragma unroll
        for (int r = 0; r < 8; ++r) {
            int row = m0 + half * 8 + r;
            if (row < M) C[(long long)row * N + col] = acc[t][r] + bs;
        }
    }
}

// ---------------- GATv2 layer 1 (2 heads x 128) ----------------

__global__ void k_score1(const float* __restrict__ xl, const float* __restrict__ xr,
                         const int* __restrict__ ei, const float* __restrict__ ea,
                         const float* __restrict__ eamean,
                         const float* __restrict__ We1, const float* __restrict__ att1,
                         float* __restrict__ sc, unsigned* __restrict__ smax,
                         int E, int Es) {
    __shared__ float sW[ED * 256];
    __shared__ float sA[256];
    for (int i = threadIdx.x; i < ED * 256; i += blockDim.x) sW[i] = We1[i];
    for (int i = threadIdx.x; i < 256; i += blockDim.x) sA[i] = att1[i];
    __syncthreads();
    const int wave = threadIdx.x >> 5, lane = threadIdx.x & 31;
    const int PER = 8;
    long long base = ((long long)blockIdx.x * 4 + wave) * PER;
    for (int j = 0; j < PER; ++j) {
        long long e = base + j;
        if (e >= Es) break;
        int s, d; const float* eap;
        if (e < E) { s = ei[e]; d = ei[E + e]; eap = ea + e * ED; }
        else       { s = d = (int)(e - E);     eap = eamean; }
        float av[ED];
        for (int k = 0; k < ED; ++k) av[k] = eap[k];
        const float* sl = xl + (long long)s * 256;
        const float* dr = xr + (long long)d * 256;
        float sum = 0.0f;
        int c0 = lane * 8;
        for (int cc = 0; cc < 8; ++cc) {
            int c = c0 + cc;
            float m = sl[c] + dr[c];
            for (int k = 0; k < ED; ++k) m += av[k] * sW[k * 256 + c];
            float lr = m > 0.0f ? m : 0.2f * m;   // leaky_relu(0.2)
            sum += lr * sA[c];
        }
        for (int msk = 1; msk <= 8; msk <<= 1) sum += __shfl_xor(sum, msk, 32);
        if ((lane & 15) == 0) {
            int hh = lane >> 4;
            sc[e * 2 + hh] = sum;
            atomicMax(&smax[(long long)d * 2 + hh], f2mono(sum));
        }
    }
}

__global__ void k_expsum1(float* __restrict__ sc, const int* __restrict__ ei,
                          const unsigned* __restrict__ smax, float* __restrict__ ssum,
                          int E, int Es) {
    long long idx = (long long)blockIdx.x * blockDim.x + threadIdx.x;
    if (idx >= (long long)Es * 2) return;
    long long e = idx >> 1; int hh = (int)(idx & 1);
    int d = (e < E) ? ei[E + e] : (int)(e - E);
    float m = mono2f(smax[(long long)d * 2 + hh]);
    float ex = expf(sc[idx] - m);
    sc[idx] = ex;
    atomicAdd(&ssum[(long long)d * 2 + hh], ex);
}

__global__ void k_agg1(const float* __restrict__ xl, const int* __restrict__ ei,
                       const float* __restrict__ sc, const float* __restrict__ ssum,
                       float* __restrict__ out, int E, int Es) {
    const int wave = threadIdx.x >> 5, lane = threadIdx.x & 31;
    const int PER = 8;
    long long base = ((long long)blockIdx.x * 4 + wave) * PER;
    for (int j = 0; j < PER; ++j) {
        long long e = base + j;
        if (e >= Es) break;
        int s, d;
        if (e < E) { s = ei[e]; d = ei[E + e]; }
        else       { s = d = (int)(e - E); }
        int hh = lane >> 4;
        float alpha = sc[e * 2 + hh] / (ssum[(long long)d * 2 + hh] + 1e-16f);
        const float* sl = xl + (long long)s * 256;
        float* orow = out + (long long)d * 256;
        int c0 = lane * 8;
        for (int cc = 0; cc < 8; ++cc)
            atomicAdd(&orow[c0 + cc], sl[c0 + cc] * alpha);
    }
}

// mean over heads + bias + layernorm + ELU  (wave per node)
__global__ void k_post1(const float* __restrict__ out, const float* __restrict__ bias,
                        const float* __restrict__ g, const float* __restrict__ be,
                        float* __restrict__ zv, int Nn) {
    const int wave = threadIdx.x >> 5, lane = threadIdx.x & 31;
    long long n = (long long)blockIdx.x * 4 + wave;
    if (n >= Nn) return;
    const float* o = out + n * 256;
    float v[4], sm = 0.0f, sq = 0.0f;
    for (int j = 0; j < 4; ++j) {
        int c = lane * 4 + j;
        float x = 0.5f * (o[c] + o[128 + c]) + bias[c];
        v[j] = x; sm += x; sq += x * x;
    }
    for (int msk = 1; msk <= 16; msk <<= 1) {
        sm += __shfl_xor(sm, msk, 32);
        sq += __shfl_xor(sq, msk, 32);
    }
    float mu = sm * (1.0f / 128.0f);
    float var = sq * (1.0f / 128.0f) - mu * mu;
    float inv = rsqrtf(var + 1e-5f);
    for (int j = 0; j < 4; ++j) {
        int c = lane * 4 + j;
        float x = (v[j] - mu) * inv * g[c] + be[c];
        zv[n * 128 + c] = x > 0.0f ? x : expm1f(x);   // ELU
    }
}

// ---------------- GATv2 layer 2 (1 head x 128) ----------------

__global__ void k_score2(const float* __restrict__ xl, const float* __restrict__ xr,
                         const int* __restrict__ ei, const float* __restrict__ ea,
                         const float* __restrict__ eamean,
                         const float* __restrict__ We2, const float* __restrict__ att2,
                         float* __restrict__ sc, unsigned* __restrict__ smax,
                         int E, int Es) {
    __shared__ float sW[ED * 128];
    __shared__ float sA[128];
    for (int i = threadIdx.x; i < ED * 128; i += blockDim.x) sW[i] = We2[i];
    for (int i = threadIdx.x; i < 128; i += blockDim.x) sA[i] = att2[i];
    __syncthreads();
    const int wave = threadIdx.x >> 5, lane = threadIdx.x & 31;
    const int PER = 8;
    long long base = ((long long)blockIdx.x * 4 + wave) * PER;
    for (int j = 0; j < PER; ++j) {
        long long e = base + j;
        if (e >= Es) break;
        int s, d; const float* eap;
        if (e < E) { s = ei[e]; d = ei[E + e]; eap = ea + e * ED; }
        else       { s = d = (int)(e - E);     eap = eamean; }
        float av[ED];
        for (int k = 0; k < ED; ++k) av[k] = eap[k];
        const float* sl = xl + (long long)s * 128;
        const float* dr = xr + (long long)d * 128;
        float sum = 0.0f;
        int c0 = lane * 4;
        for (int cc = 0; cc < 4; ++cc) {
            int c = c0 + cc;
            float m = sl[c] + dr[c];
            for (int k = 0; k < ED; ++k) m += av[k] * sW[k * 128 + c];
            float lr = m > 0.0f ? m : 0.2f * m;
            sum += lr * sA[c];
        }
        for (int msk = 1; msk <= 16; msk <<= 1) sum += __shfl_xor(sum, msk, 32);
        if (lane == 0) {
            sc[e] = sum;
            atomicMax(&smax[d], f2mono(sum));
        }
    }
}

__global__ void k_expsum2(float* __restrict__ sc, const int* __restrict__ ei,
                          const unsigned* __restrict__ smax, float* __restrict__ ssum,
                          int E, int Es) {
    long long e = (long long)blockIdx.x * blockDim.x + threadIdx.x;
    if (e >= Es) return;
    int d = (e < E) ? ei[E + e] : (int)(e - E);
    float m = mono2f(smax[d]);
    float ex = expf(sc[e] - m);
    sc[e] = ex;
    atomicAdd(&ssum[d], ex);
}

__global__ void k_agg2(const float* __restrict__ xl, const int* __restrict__ ei,
                       const float* __restrict__ sc, const float* __restrict__ ssum,
                       float* __restrict__ out, int E, int Es) {
    const int wave = threadIdx.x >> 5, lane = threadIdx.x & 31;
    const int PER = 8;
    long long base = ((long long)blockIdx.x * 4 + wave) * PER;
    for (int j = 0; j < PER; ++j) {
        long long e = base + j;
        if (e >= Es) break;
        int s, d;
        if (e < E) { s = ei[e]; d = ei[E + e]; }
        else       { s = d = (int)(e - E); }
        float alpha = sc[e] / (ssum[d] + 1e-16f);
        const float* sl = xl + (long long)s * 128;
        float* orow = out + (long long)d * 128;
        int c0 = lane * 4;
        for (int cc = 0; cc < 4; ++cc)
            atomicAdd(&orow[c0 + cc], sl[c0 + cc] * alpha);
    }
}

__global__ void k_post2(const float* __restrict__ out, const float* __restrict__ bias,
                        const float* __restrict__ g, const float* __restrict__ be,
                        float* __restrict__ zv, int Nn) {
    const int wave = threadIdx.x >> 5, lane = threadIdx.x & 31;
    long long n = (long long)blockIdx.x * 4 + wave;
    if (n >= Nn) return;
    const float* o = out + n * 128;
    float v[4], sm = 0.0f, sq = 0.0f;
    for (int j = 0; j < 4; ++j) {
        int c = lane * 4 + j;
        float x = o[c] + bias[c];
        v[j] = x; sm += x; sq += x * x;
    }
    for (int msk = 1; msk <= 16; msk <<= 1) {
        sm += __shfl_xor(sm, msk, 32);
        sq += __shfl_xor(sq, msk, 32);
    }
    float mu = sm * (1.0f / 128.0f);
    float var = sq * (1.0f / 128.0f) - mu * mu;
    float inv = rsqrtf(var + 1e-5f);
    for (int j = 0; j < 4; ++j) {
        int c = lane * 4 + j;
        float x = (v[j] - mu) * inv * g[c] + be[c];
        zv[n * 128 + c] = x > 0.0f ? x : expm1f(x);
    }
}

// ---------------- GRU with h_prev = 0 ----------------
__global__ void k_gru(const float* __restrict__ gi, const float* __restrict__ bhh,
                      float* __restrict__ h, long long total) {
    long long idx = (long long)blockIdx.x * blockDim.x + threadIdx.x;
    if (idx >= total) return;
    long long n = idx >> 7; int c = (int)(idx & 127);
    const float* gg = gi + n * 384;
    float r = 1.0f / (1.0f + expf(-(gg[c] + bhh[c])));
    float u = 1.0f / (1.0f + expf(-(gg[128 + c] + bhh[128 + c])));
    float nn = tanhf(gg[256 + c] + r * bhh[256 + c]);
    h[idx] = (1.0f - u) * nn;     // + u*0
}

// ---------------- fused edge classifier: rep(16x288) -> 128 -> 64 -> 1 ----------------
// One wave per 16-edge tile; rep/o1 staged in LDS f16; both GEMMs via WMMA.
__global__ void k_cls(const float* __restrict__ h, const int* __restrict__ ei,
                      const float* __restrict__ ea,
                      const h16* __restrict__ Wc1t, const float* __restrict__ bc1,
                      const h16* __restrict__ Wc2t, const float* __restrict__ bc2,
                      const float* __restrict__ Wc3, const float* __restrict__ bc3,
                      float* __restrict__ out, int E) {
    __shared__ h16 rep[16][288];
    __shared__ h16 o1s[16][128];
    const int lane = threadIdx.x;
    const int p = lane & 15, half = lane >> 4;
    long long e0 = (long long)blockIdx.x * 16;
    long long e = e0 + p; if (e >= E) e = E - 1;
    int node = half ? ei[E + e] : ei[e];
    const float* hr = h + (long long)node * 128;
    int cb = half * 128;
    for (int c = 0; c < 128; c += 4) {
        float4 f = *(const float4*)(hr + c);
        rep[p][cb + c + 0] = (h16)f.x; rep[p][cb + c + 1] = (h16)f.y;
        rep[p][cb + c + 2] = (h16)f.z; rep[p][cb + c + 3] = (h16)f.w;
    }
    if (!half) {
        const float* eap = ea + e * ED;
        for (int k = 0; k < ED; ++k) rep[p][256 + k] = (h16)eap[k];
        for (int k = 0; k < ED; ++k) rep[p][272 + k] = (h16)0.0f;   // K pad 272->288
    }
    __syncthreads();

    // layer 1: 16x288 @ 288x128
    v8f acc[8];
    for (int t = 0; t < 8; ++t)
        for (int r = 0; r < 8; ++r) acc[t][r] = 0.0f;
    for (int k0 = 0; k0 < 288; k0 += 32) {
        const h16* ap = &rep[p][k0 + half * 8];
        v16h a = cat8(*(const v8h*)ap, *(const v8h*)(ap + 16));
        #pragma unroll
        for (int t = 0; t < 8; ++t) {
            const h16* wp = Wc1t + (long long)(t * 16 + p) * 288 + k0 + half * 8;
            v16h b = cat8(*(const v8h*)wp, *(const v8h*)(wp + 16));
            acc[t] = wmma16(a, b, acc[t]);
        }
    }
    __syncthreads();
    #pragma unroll
    for (int t = 0; t < 8; ++t) {
        float bs = bc1[t * 16 + p];
        #pragma unroll
        for (int r = 0; r < 8; ++r) {
            float v = acc[t][r] + bs;
            o1s[half * 8 + r][t * 16 + p] = (h16)(v > 0.0f ? v : 0.0f);
        }
    }
    __syncthreads();

    // layer 2: 16x128 @ 128x64
    v8f a2[4];
    for (int t = 0; t < 4; ++t)
        for (int r = 0; r < 8; ++r) a2[t][r] = 0.0f;
    for (int k0 = 0; k0 < 128; k0 += 32) {
        const h16* ap = &o1s[p][k0 + half * 8];
        v16h a = cat8(*(const v8h*)ap, *(const v8h*)(ap + 16));
        #pragma unroll
        for (int t = 0; t < 4; ++t) {
            const h16* wp = Wc2t + (long long)(t * 16 + p) * 128 + k0 + half * 8;
            v16h b = cat8(*(const v8h*)wp, *(const v8h*)(wp + 16));
            a2[t] = wmma16(a, b, a2[t]);
        }
    }

    // layer 3: relu then 64 -> 1 GEMV with shuffle reduction
    float part[8];
    for (int r = 0; r < 8; ++r) part[r] = 0.0f;
    #pragma unroll
    for (int t = 0; t < 4; ++t) {
        float w3 = Wc3[t * 16 + p];
        float bs = bc2[t * 16 + p];
        #pragma unroll
        for (int r = 0; r < 8; ++r) {
            float v = a2[t][r] + bs;
            part[r] += (v > 0.0f ? v : 0.0f) * w3;
        }
    }
    for (int msk = 1; msk <= 8; msk <<= 1)
        for (int r = 0; r < 8; ++r) part[r] += __shfl_xor(part[r], msk, 32);
    if (p == 0) {
        float b3 = bc3[0];
        for (int r = 0; r < 8; ++r) {
            long long eo = e0 + half * 8 + r;
            if (eo < E) out[eo] = part[r] + b3;
        }
    }
}

// ---------------- host launch ----------------

static inline int cdiv(long long a, long long b) { return (int)((a + b - 1) / b); }

extern "C" void kernel_launch(void* const* d_in, const int* in_sizes, int n_in,
                              void* d_out, int out_size, void* d_ws, size_t ws_size,
                              hipStream_t stream) {
    const float* x    = (const float*)d_in[0];
    const int*   ei   = (const int*)d_in[1];
    const float* ea   = (const float*)d_in[2];
    const float* Wl1  = (const float*)d_in[3];  const float* bl1 = (const float*)d_in[4];
    const float* Wr1  = (const float*)d_in[5];  const float* br1 = (const float*)d_in[6];
    const float* We1  = (const float*)d_in[7];  const float* att1 = (const float*)d_in[8];
    const float* bias1= (const float*)d_in[9];  const float* g1 = (const float*)d_in[10];
    const float* be1  = (const float*)d_in[11];
    const float* Wl2  = (const float*)d_in[12]; const float* bl2 = (const float*)d_in[13];
    const float* Wr2  = (const float*)d_in[14]; const float* br2 = (const float*)d_in[15];
    const float* We2  = (const float*)d_in[16]; const float* att2 = (const float*)d_in[17];
    const float* bias2= (const float*)d_in[18]; const float* g2 = (const float*)d_in[19];
    const float* be2  = (const float*)d_in[20];
    const float* W_ih = (const float*)d_in[21];
    const float* b_ih = (const float*)d_in[23]; const float* b_hh = (const float*)d_in[24];
    const float* Wc1  = (const float*)d_in[25]; const float* bc1 = (const float*)d_in[26];
    const float* Wc2  = (const float*)d_in[27]; const float* bc2 = (const float*)d_in[28];
    const float* Wc3  = (const float*)d_in[29]; const float* bc3 = (const float*)d_in[30];

    const int Nn = in_sizes[0] / 128;
    const int E  = in_sizes[1] / 2;
    const int Es = E + Nn;

    // workspace layout (256B aligned slices)
    char* w = (char*)d_ws;
    auto alloc = [&](size_t bytes) -> char* {
        char* p = w; w += (bytes + 255) & ~(size_t)255; return p;
    };
    h16* Wl1t = (h16*)alloc((size_t)256 * 128 * 2);
    h16* Wr1t = (h16*)alloc((size_t)256 * 128 * 2);
    h16* Wl2t = (h16*)alloc((size_t)128 * 128 * 2);
    h16* Wr2t = (h16*)alloc((size_t)128 * 128 * 2);
    h16* Wiht = (h16*)alloc((size_t)384 * 128 * 2);
    h16* Wc1t = (h16*)alloc((size_t)128 * 288 * 2);
    h16* Wc2t = (h16*)alloc((size_t)64 * 128 * 2);
    float*    eamean = (float*)alloc(16 * 4);
    unsigned* smax1  = (unsigned*)alloc((size_t)Nn * 2 * 4);
    float*    ssum1  = (float*)alloc((size_t)Nn * 2 * 4);
    float*    sc1    = (float*)alloc((size_t)Es * 2 * 4);
    float*    bufA   = (float*)alloc((size_t)Nn * 512 * 4);  // xl1|xr1, later gi|h
    float*    out1   = (float*)alloc((size_t)Nn * 256 * 4);  // later out2
    float*    zv1    = (float*)alloc((size_t)Nn * 128 * 4);
    float*    zv2    = (float*)alloc((size_t)Nn * 128 * 4);
    float*    xl2    = (float*)alloc((size_t)Nn * 128 * 4);
    float*    xr2    = (float*)alloc((size_t)Nn * 128 * 4);
    float* xl1 = bufA;
    float* xr1 = bufA + (size_t)Nn * 256;
    float* gi  = bufA;                           // reuse after layer 1
    float* hb  = bufA + (size_t)Nn * 384;
    unsigned* smax2 = smax1; float* ssum2 = ssum1;
    float* sc2 = sc1; float* out2 = out1;

    // 1) weight prep (fp32 -> f16, transposed to [N][K])
    k_tconv<<<cdiv(256 * 128, 256), 256, 0, stream>>>(Wl1, Wl1t, 128, 256, 128, 1);
    k_tconv<<<cdiv(256 * 128, 256), 256, 0, stream>>>(Wr1, Wr1t, 128, 256, 128, 1);
    k_tconv<<<cdiv(128 * 128, 256), 256, 0, stream>>>(Wl2, Wl2t, 128, 128, 128, 1);
    k_tconv<<<cdiv(128 * 128, 256), 256, 0, stream>>>(Wr2, Wr2t, 128, 128, 128, 1);
    k_tconv<<<cdiv(384 * 128, 256), 256, 0, stream>>>(W_ih, Wiht, 128, 384, 128, 0);
    k_tconv<<<cdiv(128 * 288, 256), 256, 0, stream>>>(Wc1, Wc1t, 272, 128, 288, 1);
    k_tconv<<<cdiv(64 * 128, 256), 256, 0, stream>>>(Wc2, Wc2t, 128, 64, 128, 1);
    k_eamean<<<1, 256, 0, stream>>>(ea, eamean, E);

    // 2) layer 1
    k_fill_u32<<<cdiv((long long)Nn * 2, 256), 256, 0, stream>>>(smax1, MONO_NEG_INF, (long long)Nn * 2);
    k_fill_u32<<<cdiv((long long)Nn * 2, 256), 256, 0, stream>>>((unsigned*)ssum1, 0u, (long long)Nn * 2);
    k_fill_u32<<<cdiv((long long)Nn * 256, 256), 256, 0, stream>>>((unsigned*)out1, 0u, (long long)Nn * 256);
    k_gemm<<<dim3(cdiv(Nn, 16), 4), 32, 0, stream>>>(x, Wl1t, bl1, xl1, Nn, 128, 256);
    k_gemm<<<dim3(cdiv(Nn, 16), 4), 32, 0, stream>>>(x, Wr1t, br1, xr1, Nn, 128, 256);
    k_score1<<<cdiv(Es, 32), 128, 0, stream>>>(xl1, xr1, ei, ea, eamean, We1, att1, sc1, smax1, E, Es);
    k_expsum1<<<cdiv((long long)Es * 2, 256), 256, 0, stream>>>(sc1, ei, smax1, ssum1, E, Es);
    k_agg1<<<cdiv(Es, 32), 128, 0, stream>>>(xl1, ei, sc1, ssum1, out1, E, Es);
    k_post1<<<cdiv(Nn, 4), 128, 0, stream>>>(out1, bias1, g1, be1, zv1, Nn);

    // 3) layer 2
    k_gemm<<<dim3(cdiv(Nn, 16), 2), 32, 0, stream>>>(zv1, Wl2t, bl2, xl2, Nn, 128, 128);
    k_gemm<<<dim3(cdiv(Nn, 16), 2), 32, 0, stream>>>(zv1, Wr2t, br2, xr2, Nn, 128, 128);
    k_fill_u32<<<cdiv(Nn, 256), 256, 0, stream>>>(smax2, MONO_NEG_INF, Nn);
    k_fill_u32<<<cdiv(Nn, 256), 256, 0, stream>>>((unsigned*)ssum2, 0u, Nn);
    k_fill_u32<<<cdiv((long long)Nn * 128, 256), 256, 0, stream>>>((unsigned*)out2, 0u, (long long)Nn * 128);
    k_score2<<<cdiv(Es, 32), 128, 0, stream>>>(xl2, xr2, ei, ea, eamean, We2, att2, sc2, smax2, E, Es);
    k_expsum2<<<cdiv(Es, 256), 256, 0, stream>>>(sc2, ei, smax2, ssum2, E, Es);
    k_agg2<<<cdiv(Es, 32), 128, 0, stream>>>(xl2, ei, sc2, ssum2, out2, E, Es);
    k_post2<<<cdiv(Nn, 4), 128, 0, stream>>>(out2, bias2, g2, be2, zv2, Nn);

    // 4) GRU (h_prev = 0)
    k_gemm<<<dim3(cdiv(Nn, 16), 6), 32, 0, stream>>>(zv2, Wiht, b_ih, gi, Nn, 128, 384);
    k_gru<<<cdiv((long long)Nn * 128, 256), 256, 0, stream>>>(gi, b_hh, hb, (long long)Nn * 128);

    // 5) fused edge classifier on original edges
    k_cls<<<cdiv(E, 16), 32, 0, stream>>>(hb, ei, ea, Wc1t, bc1, Wc2t, bc2, Wc3, bc3,
                                          (float*)d_out, E);
}
